// CasualSelfAttention_10179072491728
// MI455X (gfx1250) — matrix operand order repaired and checked by reading, not compiled, not run on hardware
//
#include <hip/hip_runtime.h>

// Problem constants (match the reference)
#define BB 4
#define TT 2048
#define CC 1024
#define HH 16
#define HD 64
#define N3 (3 * CC)

typedef __attribute__((ext_vector_type(16))) __bf16 v16bf;
typedef __attribute__((ext_vector_type(8)))  __bf16 v8bf;
typedef __attribute__((ext_vector_type(8)))  float  v8f;

union FragA {
    v16bf v;
    v8bf  h[2];
};

static __device__ inline v8f zero_v8f() {
    v8f z = {0.f, 0.f, 0.f, 0.f, 0.f, 0.f, 0.f, 0.f};
    return z;
}

static __device__ inline v8f wmma_bf16(v16bf a, v16bf b, v8f c) {
    // D = A(16x32 bf16) * B(32x16 bf16) + C(16x16 f32)
    return __builtin_amdgcn_wmma_f32_16x16x32_bf16(
        /*neg_a=*/false, a, /*neg_b=*/false, b,
        /*c_mod=*/(short)0, c, /*reuse_a=*/false, /*reuse_b=*/false);
}

// Async global->LDS copy of 16 bytes (bf16 data, no conversion needed).
// Uses the CDNA5 async path (ASYNCcnt) so the copy bypasses VGPRs.
static __device__ inline void async_copy_b128(void* lds_dst, const void* gsrc) {
    const unsigned lds = (unsigned)(uintptr_t)lds_dst;       // low 32 bits = LDS addr
    const unsigned long long ga = (unsigned long long)(uintptr_t)gsrc;
    asm volatile("global_load_async_to_lds_b128 %0, %1, off"
                 :: "v"(lds), "v"(ga) : "memory");
}
static __device__ inline void async_copy_b128_ofs16(void* lds_dst, const void* gsrc) {
    const unsigned lds = (unsigned)(uintptr_t)lds_dst;
    const unsigned long long ga = (unsigned long long)(uintptr_t)gsrc;
    // INST_OFFSET applies to both LDS and global addresses (ISA 08 §4.4)
    asm volatile("global_load_async_to_lds_b128 %0, %1, off offset:16"
                 :: "v"(lds), "v"(ga) : "memory");
}

// ---------------------------------------------------------------------------
// Kernel 1: qkv = x @ W_attn + b_attn, scattered into Q[B,H,T,HD] (pre-scaled
// by 1/sqrt(HD)), K[B,H,T,HD], and V transposed Vt[B,H,HD,T], all bf16.
// Block: 256 threads (8 waves). Tile: 128(M) x 128(N), K-step 32.
// Wave grid 4(M) x 2(N); each wave computes 2x4 subtiles of 16x16.
// (f32 inputs must pass through VGPRs for bf16 conversion -> no async here.)
// ---------------------------------------------------------------------------
__global__ __launch_bounds__(256) void qkv_gemm(
        const float* __restrict__ x, const float* __restrict__ Wa,
        const float* __restrict__ ba,
        __bf16* __restrict__ Qb, __bf16* __restrict__ Kb, __bf16* __restrict__ Vt)
{
    __shared__ alignas(128) __bf16 la[128][32];   // A tile, row-major [m][k]
    __shared__ alignas(128) __bf16 lb[128][32];   // B tile, n-major  [n][k]

    const int tid    = threadIdx.x;
    const int lane   = tid & 31;
    const int wave   = tid >> 5;
    const int laneHi = lane >> 4;
    const int lane16 = lane & 15;
    const int wm     = wave & 3;   // 4 M-groups of 32 rows
    const int wn     = wave >> 2;  // 2 N-groups of 64 cols
    const int mbase  = blockIdx.y * 128;
    const int nbase  = blockIdx.x * 128;

    v8f acc[2][4];
#pragma unroll
    for (int mm = 0; mm < 2; ++mm)
#pragma unroll
        for (int nn = 0; nn < 4; ++nn) acc[mm][nn] = zero_v8f();

    const int ar = tid >> 1, ac = (tid & 1) * 16;  // A staging: 128 rows x 32
    const int br = tid >> 3, bc = (tid & 7) * 16;  // B staging: 32 rows x 128

    for (int kb = 0; kb < CC; kb += 32) {
        __syncthreads();
        // Stage A: 16 consecutive f32 -> bf16
        {
            const float4* src =
                reinterpret_cast<const float4*>(x + (size_t)(mbase + ar) * CC + kb + ac);
#pragma unroll
            for (int i = 0; i < 4; ++i) {
                float4 f = src[i];
                la[ar][ac + 4 * i + 0] = (__bf16)f.x;
                la[ar][ac + 4 * i + 1] = (__bf16)f.y;
                la[ar][ac + 4 * i + 2] = (__bf16)f.z;
                la[ar][ac + 4 * i + 3] = (__bf16)f.w;
            }
        }
        // Stage B transposed into [n][k]
        {
            const float4* src =
                reinterpret_cast<const float4*>(Wa + (size_t)(kb + br) * N3 + nbase + bc);
#pragma unroll
            for (int i = 0; i < 4; ++i) {
                float4 f = src[i];
                lb[bc + 4 * i + 0][br] = (__bf16)f.x;
                lb[bc + 4 * i + 1][br] = (__bf16)f.y;
                lb[bc + 4 * i + 2][br] = (__bf16)f.z;
                lb[bc + 4 * i + 3][br] = (__bf16)f.w;
            }
        }
        __syncthreads();

        // A fragments (16x32): lanes 0-15 row m, K 0..15; lanes 16-31 K 16..31
        FragA af[2];
#pragma unroll
        for (int mm = 0; mm < 2; ++mm) {
            const int row = wm * 32 + mm * 16 + lane16;
            const int k0  = laneHi * 16;
            af[mm].h[0] = *reinterpret_cast<const v8bf*>(&la[row][k0]);
            af[mm].h[1] = *reinterpret_cast<const v8bf*>(&la[row][k0 + 8]);
        }
        // B fragments (32x16): lane holds column N, contiguous 16 K values
#pragma unroll
        for (int nn = 0; nn < 4; ++nn) {
            const int col = wn * 64 + nn * 16 + lane16;
            const int k0  = laneHi * 16;
            v16bf bfv = *reinterpret_cast<const v16bf*>(&lb[col][k0]);
#pragma unroll
            for (int mm = 0; mm < 2; ++mm)
                acc[mm][nn] = wmma_bf16(af[mm].v, bfv, acc[mm][nn]);
        }
    }

    // Epilogue: bias, split into Q (scaled), K, V^T
#pragma unroll
    for (int mm = 0; mm < 2; ++mm) {
#pragma unroll
        for (int nn = 0; nn < 4; ++nn) {
            const int n    = nbase + wn * 64 + nn * 16 + lane16;
            const float bias = ba[n];
            const int mat = n >> 10;       // 0=Q, 1=K, 2=V (uniform per 16-wide tile)
            const int c   = n & (CC - 1);
            const int h   = c >> 6;
            const int d   = c & (HD - 1);
#pragma unroll
            for (int r = 0; r < 8; ++r) {
                const int m  = mbase + wm * 32 + mm * 16 + r + laneHi * 8;
                const int b  = m >> 11;        // T = 2048
                const int t  = m & (TT - 1);
                const int bh = b * HH + h;
                const float val = acc[mm][nn][r] + bias;
                if (mat == 0)
                    Qb[((size_t)bh * TT + t) * HD + d] = (__bf16)(val * 0.125f);
                else if (mat == 1)
                    Kb[((size_t)bh * TT + t) * HD + d] = (__bf16)val;
                else
                    Vt[((size_t)bh * HD + d) * TT + t] = (__bf16)val;
            }
        }
    }
}

// ---------------------------------------------------------------------------
// Kernel 2: causal flash attention. Grid: (T/128, B*H). 8 waves, 16 queries
// per wave, streaming 32-key blocks with online softmax. K/V blocks are
// double-buffered in LDS and prefetched with global_load_async_to_lds_b128
// (ASYNCcnt) so the next block's DMA overlaps the current block's WMMAs.
// Uniform trip count per workgroup keeps the barriers safe under causality.
// ---------------------------------------------------------------------------
__global__ __launch_bounds__(256) void flash_attn(
        const __bf16* __restrict__ Qb, const __bf16* __restrict__ Kb,
        const __bf16* __restrict__ Vt, __bf16* __restrict__ Y)
{
    __shared__ alignas(128) __bf16 lk[2][32][64];   // K block [key][hd]
    __shared__ alignas(128) __bf16 lv[2][64][32];   // V block [hd][key]
    __shared__ alignas(128) __bf16 lp[8][16][32];   // per-wave P tile [q][key]

    const int tid    = threadIdx.x;
    const int lane   = tid & 31;
    const int wave   = tid >> 5;
    const int laneHi = lane >> 4;
    const int lane16 = lane & 15;
    const int bh     = blockIdx.y;
    const int qt     = blockIdx.x;
    const int qrow0  = qt * 128 + wave * 16;
    const size_t base = (size_t)bh * TT * HD;

    // Q fragments for this wave's 16 rows (HD=64 -> two K=32 fragments)
    FragA aq[2];
    {
        const __bf16* qp = Qb + base + (size_t)(qrow0 + lane16) * HD;
#pragma unroll
        for (int h2 = 0; h2 < 2; ++h2) {
            const int k0 = h2 * 32 + laneHi * 16;
            aq[h2].h[0] = *reinterpret_cast<const v8bf*>(qp + k0);
            aq[h2].h[1] = *reinterpret_cast<const v8bf*>(qp + k0 + 8);
        }
    }

    v8f O[4];
#pragma unroll
    for (int i = 0; i < 4; ++i) O[i] = zero_v8f();
    float row_max[8], row_sum[8];
#pragma unroll
    for (int r = 0; r < 8; ++r) { row_max[r] = -1e30f; row_sum[r] = 0.f; }

    const int nblocks = (qt + 1) * 4;              // keys 0 .. qt*128+127
    const int kr = tid >> 3, kc = (tid & 7) * 8;   // lk staging 32x64, 16B/thread
    const int vr = tid >> 2, vc = (tid & 3) * 8;   // lv staging 64x32, 16B/thread

    // Issue the two async copies (K block + V block) for key-block `kb`.
    auto issue_block = [&](int kb, int buf) {
        const int kbase = kb * 32;
        async_copy_b128(&lk[buf][kr][kc],
                        Kb + base + (size_t)(kbase + kr) * HD + kc);
        async_copy_b128(&lv[buf][vr][vc],
                        Vt + base + (size_t)vr * TT + kbase + vc);
    };

    issue_block(0, 0);

    for (int kb = 0; kb < nblocks; ++kb) {
        const int buf   = kb & 1;
        const int kbase = kb * 32;

        if (kb + 1 < nblocks) {
            // buf^1 was last read in iteration kb-1; the trailing barrier of
            // that iteration makes it free for the prefetch DMA.
            issue_block(kb + 1, buf ^ 1);
            // Retire only the current block's 2 copies; keep prefetch in flight.
            asm volatile("s_wait_asynccnt 2" ::: "memory");
        } else {
            asm volatile("s_wait_asynccnt 0" ::: "memory");
        }
        __syncthreads();   // all waves' DMA for `buf` has landed

        // S = Q * K^T  (two 16-key subtiles, two HD halves)
        v8f s[2] = {zero_v8f(), zero_v8f()};
#pragma unroll
        for (int ks = 0; ks < 2; ++ks)
#pragma unroll
            for (int h2 = 0; h2 < 2; ++h2) {
                v16bf bk = *reinterpret_cast<const v16bf*>(
                    &lk[buf][ks * 16 + lane16][h2 * 32 + laneHi * 16]);
                s[ks] = wmma_bf16(aq[h2].v, bk, s[ks]);
            }

        // Causal mask + online softmax (row striped across 16-lane halves)
        float sc[8];
#pragma unroll
        for (int r = 0; r < 8; ++r) {
            const int qm = qrow0 + r + laneHi * 8;
            float v0 = s[0][r];
            float v1 = s[1][r];
            if (kbase + lane16 > qm)      v0 = -1e30f;
            if (kbase + 16 + lane16 > qm) v1 = -1e30f;
            float mx = fmaxf(v0, v1);
#pragma unroll
            for (int off = 1; off < 16; off <<= 1)
                mx = fmaxf(mx, __shfl_xor(mx, off, 16));
            const float mnew = fmaxf(row_max[r], mx);
            const float p0 = __expf(v0 - mnew);
            const float p1 = __expf(v1 - mnew);
            float ps = p0 + p1;
#pragma unroll
            for (int off = 1; off < 16; off <<= 1)
                ps += __shfl_xor(ps, off, 16);
            const float resc = __expf(row_max[r] - mnew);
            row_sum[r] = row_sum[r] * resc + ps;
            row_max[r] = mnew;
            sc[r] = resc;
            lp[wave][r + laneHi * 8][lane16]      = (__bf16)p0;
            lp[wave][r + laneHi * 8][16 + lane16] = (__bf16)p1;
        }
#pragma unroll
        for (int nsub = 0; nsub < 4; ++nsub)
#pragma unroll
            for (int r = 0; r < 8; ++r) O[nsub][r] *= sc[r];

        // Wave-private LDS round trip: make P stores visible to fragment loads
        asm volatile("s_wait_dscnt 0" ::: "memory");

        // O += P * V   (A = P 16x32, B = V 32x16 per 16-wide HD subtile)
        FragA ap;
        {
            const int k0 = laneHi * 16;
            ap.h[0] = *reinterpret_cast<const v8bf*>(&lp[wave][lane16][k0]);
            ap.h[1] = *reinterpret_cast<const v8bf*>(&lp[wave][lane16][k0 + 8]);
        }
#pragma unroll
        for (int nsub = 0; nsub < 4; ++nsub) {
            v16bf bv = *reinterpret_cast<const v16bf*>(
                &lv[buf][nsub * 16 + lane16][laneHi * 16]);
            O[nsub] = wmma_bf16(ap.v, bv, O[nsub]);
        }

        __syncthreads();   // protect `buf` from the next iteration's prefetch
    }

    // Normalize and store y in [B,T,C] bf16 (C index = h*64 + hd)
    const int b = bh >> 4, h = bh & 15;
#pragma unroll
    for (int r = 0; r < 8; ++r) {
        const float inv = 1.0f / row_sum[r];
        const int t = qrow0 + r + laneHi * 8;
#pragma unroll
        for (int nsub = 0; nsub < 4; ++nsub) {
            const int col = h * HD + nsub * 16 + lane16;
            Y[((size_t)b * TT + t) * CC + col] = (__bf16)(O[nsub][r] * inv);
        }
    }
}

// ---------------------------------------------------------------------------
// Kernel 3: out = y @ W_proj + b_proj (f32 out). Same tiling as kernel 1;
// the bf16 A tile is staged with async global->LDS DMA (no VGPR round trip).
// ---------------------------------------------------------------------------
__global__ __launch_bounds__(256) void proj_gemm(
        const __bf16* __restrict__ Y, const float* __restrict__ Wp,
        const float* __restrict__ bp, float* __restrict__ out)
{
    __shared__ alignas(128) __bf16 la[128][32];
    __shared__ alignas(128) __bf16 lb[128][32];

    const int tid    = threadIdx.x;
    const int lane   = tid & 31;
    const int wave   = tid >> 5;
    const int laneHi = lane >> 4;
    const int lane16 = lane & 15;
    const int wm     = wave & 3;
    const int wn     = wave >> 2;
    const int mbase  = blockIdx.y * 128;
    const int nbase  = blockIdx.x * 128;

    v8f acc[2][4];
#pragma unroll
    for (int mm = 0; mm < 2; ++mm)
#pragma unroll
        for (int nn = 0; nn < 4; ++nn) acc[mm][nn] = zero_v8f();

    const int ar = tid >> 1, ac = (tid & 1) * 16;
    const int br = tid >> 3, bc = (tid & 7) * 16;

    for (int kb = 0; kb < CC; kb += 32) {
        __syncthreads();
        // A tile is already bf16: 32B per thread via async DMA to LDS
        {
            const __bf16* src = Y + (size_t)(mbase + ar) * CC + kb + ac;
            async_copy_b128(&la[ar][ac], src);
            async_copy_b128_ofs16(&la[ar][ac], src);
        }
        // B tile from f32 W_proj, transposed into [n][k] (needs conversion)
        {
            const float4* src =
                reinterpret_cast<const float4*>(Wp + (size_t)(kb + br) * CC + nbase + bc);
#pragma unroll
            for (int i = 0; i < 4; ++i) {
                float4 f = src[i];
                lb[bc + 4 * i + 0][br] = (__bf16)f.x;
                lb[bc + 4 * i + 1][br] = (__bf16)f.y;
                lb[bc + 4 * i + 2][br] = (__bf16)f.z;
                lb[bc + 4 * i + 3][br] = (__bf16)f.w;
            }
        }
        asm volatile("s_wait_asynccnt 0" ::: "memory");
        __syncthreads();

        FragA af[2];
#pragma unroll
        for (int mm = 0; mm < 2; ++mm) {
            const int row = wm * 32 + mm * 16 + lane16;
            const int k0  = laneHi * 16;
            af[mm].h[0] = *reinterpret_cast<const v8bf*>(&la[row][k0]);
            af[mm].h[1] = *reinterpret_cast<const v8bf*>(&la[row][k0 + 8]);
        }
#pragma unroll
        for (int nn = 0; nn < 4; ++nn) {
            const int col = wn * 64 + nn * 16 + lane16;
            const int k0  = laneHi * 16;
            v16bf bfv = *reinterpret_cast<const v16bf*>(&lb[col][k0]);
#pragma unroll
            for (int mm = 0; mm < 2; ++mm)
                acc[mm][nn] = wmma_bf16(af[mm].v, bfv, acc[mm][nn]);
        }
    }

#pragma unroll
    for (int mm = 0; mm < 2; ++mm) {
#pragma unroll
        for (int nn = 0; nn < 4; ++nn) {
            const int n = nbase + wn * 64 + nn * 16 + lane16;
            const float bias = bp[n];
#pragma unroll
            for (int r = 0; r < 8; ++r) {
                const int m = mbase + wm * 32 + mm * 16 + r + laneHi * 8;
                out[(size_t)m * CC + n] = acc[mm][nn][r] + bias;
            }
        }
    }
}

// ---------------------------------------------------------------------------
extern "C" void kernel_launch(void* const* d_in, const int* in_sizes, int n_in,
                              void* d_out, int out_size, void* d_ws, size_t ws_size,
                              hipStream_t stream)
{
    (void)in_sizes; (void)n_in; (void)out_size; (void)ws_size;
    const float* x  = (const float*)d_in[0];
    const float* Wa = (const float*)d_in[1];
    const float* ba = (const float*)d_in[2];
    const float* Wp = (const float*)d_in[3];
    const float* bp = (const float*)d_in[4];
    float* out = (float*)d_out;

    __bf16* ws = (__bf16*)d_ws;
    const size_t NPH = (size_t)BB * HH * TT * HD;   // 8,388,608 elems per tensor
    __bf16* Qb = ws;                // [B,H,T,HD], pre-scaled by 1/sqrt(HD)
    __bf16* Kb = ws + NPH;          // [B,H,T,HD]
    __bf16* Vt = ws + 2 * NPH;      // [B,H,HD,T] (transposed)
    __bf16* Yw = ws + 3 * NPH;      // [B,T,C]

    dim3 blk(256);
    qkv_gemm<<<dim3(N3 / 128, (BB * TT) / 128), blk, 0, stream>>>(x, Wa, ba, Qb, Kb, Vt);
    flash_attn<<<dim3(TT / 128, BB * HH), blk, 0, stream>>>(Qb, Kb, Vt, Yw);
    proj_gemm<<<dim3(CC / 128, (BB * TT) / 128), blk, 0, stream>>>(Yw, Wp, bp, out);
}
